// ContextAttention_62680752718203
// MI455X (gfx1250) — matrix operand order repaired and checked by reading, not compiled, not run on hardware
//
#include <hip/hip_runtime.h>

// ---------------- problem constants ----------------
constexpr int Bsz = 64;
constexpr int Nn  = 1029;
constexpr int Dd  = 768;
constexpr int Hh  = 12;
constexpr int dh  = 64;
constexpr int Kk  = 128;
constexpr int Rr  = 5;
constexpr int Mm  = 4;
constexpr int Pp  = Nn - Rr;       // 1024
constexpr int CTXR = Rr + Kk;      // 133
constexpr int MP  = 160;           // padded kv length (multiple of 32)

typedef __attribute__((ext_vector_type(16))) __bf16 v16bf;
typedef __attribute__((ext_vector_type(8)))  float  v8f;
typedef int v4i_ __attribute__((vector_size(16)));   // matches async-LDS builtin pointee

// ---------------- helpers ----------------
__device__ __forceinline__ unsigned short f2bf(float f) {
    unsigned u = __float_as_uint(f);
    unsigned r = u + 0x7FFFu + ((u >> 16) & 1u);  // round-to-nearest-even
    return (unsigned short)(r >> 16);
}

union Frag { uint4 q[2]; v16bf v; };

// A fragment: 16x32 bf16, row-major source (ldk elements per row).
// lane L: m = L%16 ; k chunks [8*(L/16), +8) and [16+8*(L/16), +8)
__device__ __forceinline__ v16bf lda_frag(const unsigned short* base, int ldk, int kk, int lane) {
    int m = lane & 15, kb = (lane >> 4) * 8;
    const unsigned short* p = base + (long long)m * ldk + kk + kb;
    Frag u;
    u.q[0] = *(const uint4*)p;
    u.q[1] = *(const uint4*)(p + 16);
    return u.v;
}

// B fragment: 32x16 bf16 taken from B^T rows (N x K, row-major).
// lane L: n = L%16 ; k chunks [8*(L/16), +8) and [16+8*(L/16), +8)
__device__ __forceinline__ v16bf ldb_frag(const unsigned short* base, int ldk, int kk, int lane) {
    int n = lane & 15, kb = (lane >> 4) * 8;
    const unsigned short* p = base + (long long)n * ldk + kk + kb;
    Frag u;
    u.q[0] = *(const uint4*)p;
    u.q[1] = *(const uint4*)(p + 16);
    return u.v;
}

__device__ __forceinline__ v8f wmma_bf16(v16bf a, v16bf b, v8f c) {
    return __builtin_amdgcn_wmma_f32_16x16x32_bf16(false, a, false, b, (short)0, c, false, false);
}

__device__ __forceinline__ v8f vzero8() { v8f z = {0.f,0.f,0.f,0.f,0.f,0.f,0.f,0.f}; return z; }

// ---- CDNA5 async global->LDS copy (ASYNCcnt path), guarded with fallback ----
#if defined(__gfx1250__) && __has_builtin(__builtin_amdgcn_global_load_async_to_lds_b128)
#define USE_ASYNC_LDS 1
#else
#define USE_ASYNC_LDS 0
#endif

__device__ __forceinline__ void cp16(const unsigned short* gsrc, unsigned short* ldst) {
#if USE_ASYNC_LDS
    __builtin_amdgcn_global_load_async_to_lds_b128(
        (__attribute__((address_space(1))) v4i_*)gsrc,
        (__attribute__((address_space(3))) v4i_*)ldst, 0, 0);
#else
    *(uint4*)ldst = *(const uint4*)gsrc;
#endif
}

__device__ __forceinline__ void cp16_wait() {
#if USE_ASYNC_LDS
#if __has_builtin(__builtin_amdgcn_s_wait_asynccnt)
    __builtin_amdgcn_s_wait_asynccnt(0);
#else
    asm volatile("s_wait_asynccnt 0" ::: "memory");
#endif
#endif
}

// ---------------- elementwise conversions ----------------
__global__ void f32_to_bf16_kernel(const float* __restrict__ in, unsigned short* __restrict__ out, long long n) {
    long long i = ((long long)blockIdx.x * blockDim.x + threadIdx.x) * 4;
    if (i + 3 < n) {
        float4 v = *(const float4*)&in[i];
        out[i+0] = f2bf(v.x); out[i+1] = f2bf(v.y);
        out[i+2] = f2bf(v.z); out[i+3] = f2bf(v.w);
    } else {
        for (int j = 0; j < 4; ++j) if (i + j < n) out[i+j] = f2bf(in[i+j]);
    }
}

__global__ void transpose_f32_to_bf16_kernel(const float* __restrict__ in, unsigned short* __restrict__ out,
                                             int rows, int cols) {
    long long t = (long long)blockIdx.x * blockDim.x + threadIdx.x;
    if (t >= (long long)rows * cols) return;
    int c = (int)(t % cols), r = (int)(t / cols);
    out[(long long)c * rows + r] = f2bf(in[t]);
}

// ---------------- CLS norm + routing ----------------
__global__ __launch_bounds__(256)
void cls_kernel(const float* __restrict__ x, const float* __restrict__ centroids,
                float* __restrict__ cls_out, int* __restrict__ idx_out, int* __restrict__ ws_idx) {
    int b = blockIdx.x, tid = threadIdx.x;
    __shared__ float red[256];
    __shared__ float cls[Dd];
    __shared__ float sims[Mm];
    const float* xb = x + (long long)b * Nn * Dd;  // row 0 = CLS
    float ss = 0.f;
    for (int i = tid; i < Dd; i += 256) { float v = xb[i]; ss += v * v; }
    red[tid] = ss; __syncthreads();
    for (int st = 128; st > 0; st >>= 1) {
        if (tid < st) red[tid] += red[tid + st];
        __syncthreads();
    }
    float inv = 1.f / fmaxf(sqrtf(red[0]), 1e-12f);
    for (int i = tid; i < Dd; i += 256) {
        float v = xb[i] * inv;
        cls[i] = v;
        cls_out[(long long)b * Dd + i] = v;
    }
    __syncthreads();
    if (tid < Mm) {
        float dsum = 0.f;
        for (int i = 0; i < Dd; ++i) dsum += cls[i] * centroids[tid * Dd + i];
        sims[tid] = dsum;
    }
    __syncthreads();
    if (tid == 0) {
        int best = 0; float bv = sims[0];
        for (int m = 1; m < Mm; ++m) if (sims[m] > bv) { bv = sims[m]; best = m; }
        idx_out[b] = best;
        ws_idx[b] = best;
    }
}

__global__ void gather_qctx_kernel(const float* __restrict__ Qb, const int* __restrict__ idx,
                                   unsigned short* __restrict__ out) {
    long long t = (long long)blockIdx.x * blockDim.x + threadIdx.x;
    long long tot = (long long)Bsz * Kk * Dd;
    if (t >= tot) return;
    int b = (int)(t / (Kk * Dd));
    long long j = t % (Kk * Dd);
    out[t] = f2bf(Qb[(long long)idx[b] * Kk * Dd + j]);
}

__global__ void copy_xreg_kernel(const float* __restrict__ x, unsigned short* __restrict__ ctx_bf) {
    long long t = (long long)blockIdx.x * blockDim.x + threadIdx.x;
    long long tot = (long long)Bsz * Rr * Dd;
    if (t >= tot) return;
    int b = (int)(t / (Rr * Dd));
    long long j = t % (Rr * Dd);
    ctx_bf[(long long)b * CTXR * Dd + j] = f2bf(x[(long long)b * Nn * Dd + j]);
}

// ------- bf16 WMMA GEMM (NT), 128x128 block tile, 8 waves, double-buffered LDS -------
// C = A(MxK) @ B'(NxK)^T ; requires N % 128 == 0, Kd % 32 == 0.
__global__ __launch_bounds__(256)
void gemm_nt_kernel(const unsigned short* __restrict__ A, int lda, long long strA,
                    const unsigned short* __restrict__ Bp, int ldb, long long strB,
                    void* __restrict__ Cv, int ldc, long long strC,
                    int M, int N, int Kd, const float* __restrict__ bias,
                    float scale, int c_bf16) {
    __shared__ __align__(16) unsigned short As[2][128 * 32];
    __shared__ __align__(16) unsigned short Bs[2][128 * 32];
    int tid = threadIdx.x, lane = tid & 31, wv = tid >> 5;
    int wr = wv & 3, wc = wv >> 2;           // wave row (4) x wave col (2)
    int m0 = blockIdx.y * 128, n0 = blockIdx.x * 128;
    const unsigned short* Ab = A + (long long)blockIdx.z * strA;
    const unsigned short* Bb = Bp + (long long)blockIdx.z * strB;

    v8f acc[2][4];
#pragma unroll
    for (int s = 0; s < 2; ++s)
#pragma unroll
        for (int t = 0; t < 4; ++t) acc[s][t] = vzero8();

    // staging: 128 rows x 32 bf16 (64B) per operand -> 512 x 16B chunks; 256 threads x 2
    auto stage = [&](int buf, int kk) {
#pragma unroll
        for (int it = 0; it < 2; ++it) {
            int u = tid + it * 256;          // 0..511
            int row = u >> 2, ch = u & 3;    // 128 rows x 4 chunks(8 bf16)
            int gm = m0 + row; gm = gm < M ? gm : (M - 1);
            cp16(&Ab[(long long)gm * lda + kk + ch * 8], &As[buf][row * 32 + ch * 8]);
            cp16(&Bb[(long long)(n0 + row) * ldb + kk + ch * 8], &Bs[buf][row * 32 + ch * 8]);
        }
    };

    stage(0, 0);
    cp16_wait();
    __syncthreads();

    for (int kk = 0; kk < Kd; kk += 32) {
        int cur = (kk >> 5) & 1;
        bool more = (kk + 32) < Kd;
        if (more) stage(cur ^ 1, kk + 32);   // prefetch next slice into other buffer

        v16bf a0 = lda_frag(&As[cur][(wr * 32) * 32], 32, 0, lane);
        v16bf a1 = lda_frag(&As[cur][(wr * 32 + 16) * 32], 32, 0, lane);
#pragma unroll
        for (int t = 0; t < 4; ++t) {
            v16bf bf = ldb_frag(&Bs[cur][(wc * 64 + t * 16) * 32], 32, 0, lane);
            acc[0][t] = wmma_bf16(a0, bf, acc[0][t]);
            acc[1][t] = wmma_bf16(a1, bf, acc[1][t]);
        }
        if (more) cp16_wait();
        __syncthreads();
    }

    long long cb = (long long)blockIdx.z * strC;
#pragma unroll
    for (int s = 0; s < 2; ++s)
#pragma unroll
        for (int t = 0; t < 4; ++t)
#pragma unroll
            for (int r = 0; r < 8; ++r) {
                int gm = m0 + wr * 32 + s * 16 + r + 8 * (lane >> 4);
                int gn = n0 + wc * 64 + t * 16 + (lane & 15);
                if (gm < M) {
                    float v = acc[s][t][r] * scale + (bias ? bias[gn] : 0.f);
                    if (c_bf16) ((unsigned short*)Cv)[cb + (long long)gm * ldc + gn] = f2bf(v);
                    else        ((float*)Cv)[cb + (long long)gm * ldc + gn] = v;
                }
            }
}

// ---------------- NN variant: C = A(MxK) @ B(KxN), B staged transposed via LDS ----------------
__global__ __launch_bounds__(128)
void gemm_nn_kernel(const unsigned short* __restrict__ A, int lda, long long strA,
                    const unsigned short* __restrict__ Bp, int ldb, long long strB,
                    void* __restrict__ Cv, int ldc, long long strC,
                    int M, int N, int Kd, float scale, int c_bf16) {
    __shared__ __align__(16) unsigned short As[64 * 32];
    __shared__ __align__(16) unsigned short Bs[64 * 32];   // stored as [n_local][k_local]
    int tid = threadIdx.x, lane = tid & 31, wv = tid >> 5;
    int m0 = blockIdx.y * 64, n0 = blockIdx.x * 64;
    const unsigned short* Ab = A + (long long)blockIdx.z * strA;
    const unsigned short* Bb = Bp + (long long)blockIdx.z * strB;
    v8f acc[4];
#pragma unroll
    for (int t = 0; t < 4; ++t) acc[t] = vzero8();

    for (int kk = 0; kk < Kd; kk += 32) {
#pragma unroll
        for (int it = 0; it < 2; ++it) {
            int u = tid + it * 128;
            int row = u >> 2, ch = u & 3;
            int gm = m0 + row; gm = gm < M ? gm : (M - 1);
            *(uint4*)&As[row * 32 + ch * 8] = *(const uint4*)&Ab[(long long)gm * lda + kk + ch * 8];
        }
#pragma unroll
        for (int it = 0; it < 2; ++it) {
            int u = tid + it * 128;           // 0..255
            int krow = u >> 3, ch = u & 7;    // 32 k-rows x 8 chunks(8 bf16 along n)
            uint4 val = *(const uint4*)&Bb[(long long)(kk + krow) * ldb + n0 + ch * 8];
            const unsigned short* pv = (const unsigned short*)&val;
#pragma unroll
            for (int e = 0; e < 8; ++e) Bs[(ch * 8 + e) * 32 + krow] = pv[e];
        }
        __syncthreads();
        v16bf af = lda_frag(&As[wv * 16 * 32], 32, 0, lane);
#pragma unroll
        for (int t = 0; t < 4; ++t) {
            v16bf bf = ldb_frag(&Bs[t * 16 * 32], 32, 0, lane);
            acc[t] = wmma_bf16(af, bf, acc[t]);
        }
        __syncthreads();
    }

    long long cb = (long long)blockIdx.z * strC;
#pragma unroll
    for (int t = 0; t < 4; ++t)
#pragma unroll
        for (int r = 0; r < 8; ++r) {
            int gm = m0 + wv * 16 + r + 8 * (lane >> 4);
            int gn = n0 + t * 16 + (lane & 15);
            if (gm < M) {
                float v = acc[t][r] * scale;
                if (c_bf16) ((unsigned short*)Cv)[cb + (long long)gm * ldc + gn] = f2bf(v);
                else        ((float*)Cv)[cb + (long long)gm * ldc + gn] = v;
            }
        }
}

// ---------------- row softmax f32 -> bf16 (context attention) ----------------
__global__ __launch_bounds__(256)
void softmax_rows_kernel(const float* __restrict__ S, unsigned short* __restrict__ P, int cols) {
    long long row = blockIdx.x;
    const float* s = S + row * cols;
    int tid = threadIdx.x;
    __shared__ float red[256];
    float mx = -3.0e38f;
    for (int c = tid; c < cols; c += 256) mx = fmaxf(mx, s[c]);
    red[tid] = mx; __syncthreads();
    for (int st = 128; st > 0; st >>= 1) { if (tid < st) red[tid] = fmaxf(red[tid], red[tid + st]); __syncthreads(); }
    mx = red[0]; __syncthreads();
    float sum = 0.f;
    for (int c = tid; c < cols; c += 256) sum += __expf(s[c] - mx);
    red[tid] = sum; __syncthreads();
    for (int st = 128; st > 0; st >>= 1) { if (tid < st) red[tid] += red[tid + st]; __syncthreads(); }
    float inv = 1.f / red[0];
    for (int c = tid; c < cols; c += 256) P[row * cols + c] = f2bf(__expf(s[c] - mx) * inv);
}

// ---------------- repack kv -> k(B,H,MP,dh) and v^T(B,H,dh,MP), zero padded ----------------
__global__ void repack_kv_kernel(const unsigned short* __restrict__ kv,
                                 unsigned short* __restrict__ kbf, unsigned short* __restrict__ vt) {
    long long t = (long long)blockIdx.x * blockDim.x + threadIdx.x;
    long long tot = (long long)Bsz * Hh * MP * dh;
    if (t >= tot) return;
    int ddc = (int)(t & (dh - 1));
    int m   = (int)((t >> 6) % MP);
    int h   = (int)(((t >> 6) / MP) % Hh);
    int b   = (int)(((t >> 6) / MP) / Hh);
    unsigned short kvv = 0, vvv = 0;
    if (m < CTXR) {
        const unsigned short* r = kv + ((long long)b * CTXR + m) * (2 * Dd);
        kvv = r[h * dh + ddc];
        vvv = r[Dd + h * dh + ddc];
    }
    kbf[(((long long)b * Hh + h) * MP + m) * dh + ddc] = kvv;
    vt [(((long long)b * Hh + h) * dh + ddc) * MP + m] = vvv;
}

// ---------------- fused main attention: per (b,h,16-row tile) ----------------
__global__ __launch_bounds__(32)
void attn_kernel(const unsigned short* __restrict__ qbf, const unsigned short* __restrict__ kbf,
                 const unsigned short* __restrict__ vt, unsigned short* __restrict__ ybf) {
    __shared__ __align__(16) unsigned short qs[16 * dh];
    __shared__ __align__(16) float Ss[16 * MP];
    __shared__ __align__(16) unsigned short Ps[16 * MP];
    int lane = threadIdx.x;
    int n0 = blockIdx.x * 16, h = blockIdx.y, b = blockIdx.z;

    // stage 16x64 q tile (scale already folded into q projection)
#pragma unroll
    for (int it = 0; it < 4; ++it) {
        int u = lane + it * 32;          // 0..127
        int row = u >> 3, ch = u & 7;    // 16 rows x 8 chunks(8 bf16)
        int gr = n0 + row; gr = gr < Nn ? gr : (Nn - 1);
        cp16(&qbf[((long long)b * Nn + gr) * Dd + h * dh + ch * 8], &qs[row * dh + ch * 8]);
    }
    cp16_wait();
    __syncthreads();

    // S = q @ k^T  (16 x 160), k stored (MP x dh) row-major => B' rows contiguous in K
    const unsigned short* kb = kbf + ((long long)(b * Hh + h)) * MP * dh;
    v8f sacc[MP / 16];
#pragma unroll
    for (int t = 0; t < MP / 16; ++t) sacc[t] = vzero8();
#pragma unroll
    for (int kk = 0; kk < dh; kk += 32) {
        v16bf af = lda_frag(qs, dh, kk, lane);
#pragma unroll
        for (int t = 0; t < MP / 16; ++t) {
            v16bf bf = ldb_frag(kb + t * 16 * dh, dh, kk, lane);
            sacc[t] = wmma_bf16(af, bf, sacc[t]);
        }
    }
#pragma unroll
    for (int t = 0; t < MP / 16; ++t)
#pragma unroll
        for (int r = 0; r < 8; ++r)
            Ss[(r + 8 * (lane >> 4)) * MP + t * 16 + (lane & 15)] = sacc[t][r];
    __syncthreads();

    // softmax over 133 valid columns (lane pairs split each row in halves)
    {
        int row = lane & 15, half = lane >> 4;
        int c0 = half * (MP / 2);
        float mx = -3.0e38f;
        for (int c = c0; c < c0 + MP / 2; ++c)
            if (c < CTXR) mx = fmaxf(mx, Ss[row * MP + c]);
        mx = fmaxf(mx, __shfl_xor(mx, 16));
        float sum = 0.f;
        for (int c = c0; c < c0 + MP / 2; ++c) {
            float p = (c < CTXR) ? __expf(Ss[row * MP + c] - mx) : 0.f;
            Ss[row * MP + c] = p;
            sum += p;
        }
        sum += __shfl_xor(sum, 16);
        float inv = 1.f / sum;
        for (int c = c0; c < c0 + MP / 2; ++c)
            Ps[row * MP + c] = f2bf(Ss[row * MP + c] * inv);
    }
    __syncthreads();

    // O = P @ v  (16 x 64), v^T stored (dh x MP) row-major
    const unsigned short* vb = vt + ((long long)(b * Hh + h)) * dh * MP;
    v8f oacc[dh / 16];
#pragma unroll
    for (int t = 0; t < dh / 16; ++t) oacc[t] = vzero8();
#pragma unroll
    for (int kk = 0; kk < MP; kk += 32) {
        v16bf af = lda_frag(Ps, MP, kk, lane);
#pragma unroll
        for (int t = 0; t < dh / 16; ++t) {
            v16bf bf = ldb_frag(vb + t * 16 * MP, MP, kk, lane);
            oacc[t] = wmma_bf16(af, bf, oacc[t]);
        }
    }
#pragma unroll
    for (int t = 0; t < dh / 16; ++t)
#pragma unroll
        for (int r = 0; r < 8; ++r) {
            int gm = n0 + r + 8 * (lane >> 4);
            if (gm < Nn)
                ybf[((long long)b * Nn + gm) * Dd + h * dh + t * 16 + (lane & 15)] = f2bf(oacc[t][r]);
        }
}

// ---------------- workspace layout ----------------
static constexpr size_t al256(size_t x) { return (x + 255) & ~(size_t)255; }
constexpr size_t SZ_XBF   = (size_t)Bsz * Nn * Dd * 2;
constexpr size_t SZ_QBF   = (size_t)Bsz * Nn * Dd * 2;
constexpr size_t SZ_YBF   = (size_t)Bsz * Nn * Dd * 2;
constexpr size_t SZ_SCORE = (size_t)Bsz * Kk * Pp * 4;
constexpr size_t SZ_ATTN  = (size_t)Bsz * Kk * Pp * 2;
constexpr size_t SZ_QCTX  = (size_t)Bsz * Kk * Dd * 2;
constexpr size_t SZ_CTX   = (size_t)Bsz * CTXR * Dd * 2;
constexpr size_t SZ_KV    = (size_t)Bsz * CTXR * 2 * Dd * 2;
constexpr size_t SZ_KBF   = (size_t)Bsz * Hh * MP * dh * 2;
constexpr size_t SZ_VT    = (size_t)Bsz * Hh * dh * MP * 2;
constexpr size_t SZ_WQT   = (size_t)Dd * Dd * 2;
constexpr size_t SZ_WCT   = (size_t)Dd * 2 * Dd * 2;
constexpr size_t SZ_WOT   = (size_t)Dd * Dd * 2;

constexpr size_t OFF_XBF   = 0;
constexpr size_t OFF_QBF   = al256(OFF_XBF + SZ_XBF);
constexpr size_t OFF_YBF   = al256(OFF_QBF + SZ_QBF);
constexpr size_t OFF_SCORE = al256(OFF_YBF + SZ_YBF);
constexpr size_t OFF_ATTN  = al256(OFF_SCORE + SZ_SCORE);
constexpr size_t OFF_QCTX  = al256(OFF_ATTN + SZ_ATTN);
constexpr size_t OFF_CTX   = al256(OFF_QCTX + SZ_QCTX);
constexpr size_t OFF_KV    = al256(OFF_CTX + SZ_CTX);
constexpr size_t OFF_KBF   = al256(OFF_KV + SZ_KV);
constexpr size_t OFF_VT    = al256(OFF_KBF + SZ_KBF);
constexpr size_t OFF_WQT   = al256(OFF_VT + SZ_VT);
constexpr size_t OFF_WCT   = al256(OFF_WQT + SZ_WQT);
constexpr size_t OFF_WOT   = al256(OFF_WCT + SZ_WCT);
constexpr size_t OFF_IDX   = al256(OFF_WOT + SZ_WOT);

// ---------------- launch ----------------
extern "C" void kernel_launch(void* const* d_in, const int* in_sizes, int n_in,
                              void* d_out, int out_size, void* d_ws, size_t ws_size,
                              hipStream_t stream) {
    const float* x         = (const float*)d_in[0];
    const float* Q_banks   = (const float*)d_in[1];
    const float* Wq        = (const float*)d_in[2];
    const float* Wctx      = (const float*)d_in[3];
    const float* bctx      = (const float*)d_in[4];
    const float* Wout      = (const float*)d_in[5];
    const float* bout      = (const float*)d_in[6];
    const float* centroids = (const float*)d_in[7];

    char* ws = (char*)d_ws;
    unsigned short* x_bf    = (unsigned short*)(ws + OFF_XBF);
    unsigned short* q_bf    = (unsigned short*)(ws + OFF_QBF);
    unsigned short* y_bf    = (unsigned short*)(ws + OFF_YBF);
    float*          scores  = (float*)        (ws + OFF_SCORE);
    unsigned short* attn_bf = (unsigned short*)(ws + OFF_ATTN);
    unsigned short* qctx_bf = (unsigned short*)(ws + OFF_QCTX);
    unsigned short* ctx_bf  = (unsigned short*)(ws + OFF_CTX);
    unsigned short* kv_bf   = (unsigned short*)(ws + OFF_KV);
    unsigned short* k_bf    = (unsigned short*)(ws + OFF_KBF);
    unsigned short* v_t     = (unsigned short*)(ws + OFF_VT);
    unsigned short* WqT     = (unsigned short*)(ws + OFF_WQT);
    unsigned short* WctxT   = (unsigned short*)(ws + OFF_WCT);
    unsigned short* WoutT   = (unsigned short*)(ws + OFF_WOT);
    int*            ws_idx  = (int*)           (ws + OFF_IDX);

    float* dout    = (float*)d_out;
    float* cls_out = dout + (size_t)Bsz * Nn * Dd;
    int*   idx_out = (int*)(dout + (size_t)Bsz * Nn * Dd + (size_t)Bsz * Dd);

    // 1) x -> bf16
    {
        long long n = (long long)Bsz * Nn * Dd;
        int blocks = (int)((n / 4 + 255) / 256);
        f32_to_bf16_kernel<<<blocks, 256, 0, stream>>>(x, x_bf, n);
    }
    // 2) transpose weights -> bf16 (so all weight GEMMs are NT)
    transpose_f32_to_bf16_kernel<<<(Dd * Dd + 255) / 256, 256, 0, stream>>>(Wq, WqT, Dd, Dd);
    transpose_f32_to_bf16_kernel<<<(Dd * 2 * Dd + 255) / 256, 256, 0, stream>>>(Wctx, WctxT, Dd, 2 * Dd);
    transpose_f32_to_bf16_kernel<<<(Dd * Dd + 255) / 256, 256, 0, stream>>>(Wout, WoutT, Dd, Dd);
    // 3) cls / routing
    cls_kernel<<<Bsz, 256, 0, stream>>>(x, centroids, cls_out, idx_out, ws_idx);
    // 4) gather Q_banks[idx] -> bf16
    gather_qctx_kernel<<<(int)(((long long)Bsz * Kk * Dd + 255) / 256), 256, 0, stream>>>(Q_banks, ws_idx, qctx_bf);
    // 5) context scores: (B) 128x1024 = qctx(128x768) @ xp(1024x768)^T   [f32 out]
    gemm_nt_kernel<<<dim3(Pp / 128, (Kk + 127) / 128, Bsz), 256, 0, stream>>>(
        qctx_bf, Dd, (long long)Kk * Dd,
        x_bf + (long long)Rr * Dd, Dd, (long long)Nn * Dd,
        scores, Pp, (long long)Kk * Pp,
        Kk, Pp, Dd, nullptr, 1.0f, 0);
    // 6) softmax rows -> attn bf16
    softmax_rows_kernel<<<Bsz * Kk, 256, 0, stream>>>(scores, attn_bf, Pp);
    // 7) ctx rows 0..4 = xreg
    copy_xreg_kernel<<<(int)(((long long)Bsz * Rr * Dd + 255) / 256), 256, 0, stream>>>(x, ctx_bf);
    // 8) ctx_p = attn(128x1024) @ xp(1024x768)  [NN, bf16 out into ctx rows 5..132]
    gemm_nn_kernel<<<dim3(Dd / 64, (Kk + 63) / 64, Bsz), 128, 0, stream>>>(
        attn_bf, Pp, (long long)Kk * Pp,
        x_bf + (long long)Rr * Dd, Dd, (long long)Nn * Dd,
        ctx_bf + (long long)Rr * Dd, Dd, (long long)CTXR * Dd,
        Kk, Dd, Pp, 1.0f, 1);
    // 9) kv = ctx(133x768) @ Wctx(768x1536) + bctx  [bf16 out]
    gemm_nt_kernel<<<dim3(2 * Dd / 128, (CTXR + 127) / 128, Bsz), 256, 0, stream>>>(
        ctx_bf, Dd, (long long)CTXR * Dd,
        WctxT, Dd, 0,
        kv_bf, 2 * Dd, (long long)CTXR * 2 * Dd,
        CTXR, 2 * Dd, Dd, bctx, 1.0f, 1);
    // 10) repack kv -> k(B,H,160,64), v^T(B,H,64,160) with zero pad
    repack_kv_kernel<<<(int)(((long long)Bsz * Hh * MP * dh + 255) / 256), 256, 0, stream>>>(kv_bf, k_bf, v_t);
    // 11) q = x @ Wq, scaled by 1/sqrt(64), bf16 out (flattened M = B*N)
    gemm_nt_kernel<<<dim3(Dd / 128, (Bsz * Nn + 127) / 128, 1), 256, 0, stream>>>(
        x_bf, Dd, 0,
        WqT, Dd, 0,
        q_bf, Dd, 0,
        Bsz * Nn, Dd, Dd, nullptr, 0.125f, 1);
    // 12) fused attention -> y bf16
    attn_kernel<<<dim3((Nn + 15) / 16, Hh, Bsz), 32, 0, stream>>>(q_bf, k_bf, v_t, y_bf);
    // 13) out = y @ Wout + bout  [f32 into d_out]
    gemm_nt_kernel<<<dim3(Dd / 128, (Bsz * Nn + 127) / 128, 1), 256, 0, stream>>>(
        y_bf, Dd, 0,
        WoutT, Dd, 0,
        dout, Dd, 0,
        Bsz * Nn, Dd, Dd, bout, 1.0f, 0);
}
